// StructureModule_73950746902797
// MI455X (gfx1250) — compile-verified, matrix operand order checked by reading, not run on hardware
//
#include <hip/hip_runtime.h>
#include <math.h>

// ---------------------------------------------------------------------------
// StructureModule (IPA + EGNN + FFN, 2 layers) for gfx1250 (MI455X, wave32).
// Everything is L2-resident (< 10 MB working set vs 192 MB L2), so all GEMMs
// feed v_wmma_f32_16x16x32_f16 directly from global memory with b128 fragment
// loads: weights are pre-transposed f16 [n][k] (B-fragment layout), and every
// activation consumed as GEMM-A keeps an f16 mirror. No LDS staging, no
// predicated scalar loads.
// EGNN e0 is linear in [fi,fj,dist] -> split into node GEMMs Pi/Pj; only the
// nonlinear per-edge 128x128 GEMMs (e1, c0) stream over N^2 edges, fused.
// Frames are identity in the reference, so point rotations are skipped.
// ---------------------------------------------------------------------------

#define NRES 384
#define DIMC 256
#define HC   8
#define DH   32
#define PQC  4
#define PVC  8
#define HIDD 128

typedef _Float16 half_t;
typedef __attribute__((ext_vector_type(16))) _Float16 v16h;
typedef __attribute__((ext_vector_type(8)))  float    v8f;

enum { ACT_NONE = 0, ACT_GELU = 1, ACT_SILU = 2, ACT_SIGMOID = 3 };

__device__ __forceinline__ float act_apply(float x, int act) {
  switch (act) {
    case ACT_GELU:    return 0.5f * x * (1.0f + erff(x * 0.70710678118654752f));
    case ACT_SILU:    return x / (1.0f + __expf(-x));
    case ACT_SIGMOID: return 1.0f / (1.0f + __expf(-x));
    default:          return x;
  }
}

// ---------------------------------------------------------------------------
// WMMA GEMM, all-global-fragment version.
//   C[M][Nd](f32) = act(Ah[M][K](f16) @ Wt[Npad][K](f16)^T + bias) (+ res)
//   Ch (optional) = f16 mirror of C for the next GEMM in the chain.
// Requirements (all call sites satisfy): M % 64 == 0, K % 32 == 0,
// Wt padded with zero rows to a multiple of 32.
// Block = 256 threads = 8 waves (4 along M x 2 along N); wave tile 16x16.
// Inner loop: 4 x global_load_b128 + 1 x v_wmma_f32_16x16x32_f16.
// ---------------------------------------------------------------------------
__global__ void gemm_kernel(const half_t* __restrict__ Ah, const half_t* __restrict__ Wt,
                            const float* __restrict__ bias, const float* __restrict__ res,
                            float* __restrict__ C, half_t* __restrict__ Ch,
                            int M, int K, int Nd, int act) {
  const int tid  = threadIdx.x;
  const int lane = tid & 31;
  const int wave = tid >> 5;
  const int wm   = wave & 3;
  const int wn   = wave >> 2;
  const int m0   = blockIdx.x * 64;
  const int n0   = blockIdx.y * 32;

  const int r    = lane & 15;
  const int koff = (lane >> 4) * 8;     // lanes 16..31 hold K+8 per 16-bit layout
  const half_t* arow = Ah + (size_t)(m0 + wm * 16 + r) * K;
  const half_t* brow = Wt + (size_t)(n0 + wn * 16 + r) * K;

  v8f acc = {};
#pragma unroll 2
  for (int k0 = 0; k0 < K; k0 += 32) {
    v16h av, bv;
    ((uint4*)&av)[0] = *(const uint4*)(arow + k0 + koff);
    ((uint4*)&av)[1] = *(const uint4*)(arow + k0 + 16 + koff);
    ((uint4*)&bv)[0] = *(const uint4*)(brow + k0 + koff);
    ((uint4*)&bv)[1] = *(const uint4*)(brow + k0 + 16 + koff);
    acc = __builtin_amdgcn_wmma_f32_16x16x32_f16(false, av, false, bv, (short)0, acc,
                                                 false, false);
  }

  // D layout: col = lane&15 (+wn*16), row = v + (lane>=16 ? 8 : 0) (+wm*16)
  const int ncol = n0 + wn * 16 + r;
  if (ncol < Nd) {
    const float b = bias ? bias[ncol] : 0.0f;
    const int mbase = m0 + wm * 16 + ((lane >> 4) ? 8 : 0);
#pragma unroll
    for (int v = 0; v < 8; ++v) {
      const int mrow = mbase + v;
      float x = acc[v] + b;
      x = act_apply(x, act);
      if (res) x += res[(size_t)mrow * Nd + ncol];
      C[(size_t)mrow * Nd + ncol] = x;
      if (Ch) Ch[(size_t)mrow * Nd + ncol] = (half_t)x;
    }
  }
}

// Transpose + f32->f16 + N-pad a weight W[K][N] into Wt[Npad][K] so WMMA
// B-fragments are two contiguous b128 loads per K-step.
__global__ void trconv_kernel(const float* __restrict__ W, half_t* __restrict__ Wt,
                              int K, int N, int Npad) {
  const int idx = blockIdx.x * 256 + threadIdx.x;
  if (idx < Npad * K) {
    const int n = idx / K, k = idx % K;
    Wt[idx] = (n < N) ? (half_t)W[(size_t)k * N + n] : (half_t)0.0f;
  }
}

// ---------------------------------------------------------------------------
// IPA attention: one block per (n, h); 384 threads (one per key m).
// Writes attn_feat / attn_pts (f16) directly into the 448-wide concat buffer
// that feeds the output-projection GEMM.
// ---------------------------------------------------------------------------
__global__ void ipa_attn_kernel(const float* __restrict__ Q, const float* __restrict__ K,
                                const float* __restrict__ V, const float* __restrict__ qp,
                                const float* __restrict__ kp, const float* __restrict__ vp,
                                const float* __restrict__ w_c, const float* __restrict__ w_l,
                                half_t* __restrict__ cat448h) {
  const int n = blockIdx.x, h = blockIdx.y, m = threadIdx.x;
  __shared__ float wsm[NRES];

  const float* qrow = Q + (size_t)n * DIMC + h * DH;
  const float* krow = K + (size_t)m * DIMC + h * DH;
  float lf = 0.0f;
#pragma unroll
  for (int d = 0; d < DH; ++d) lf += qrow[d] * krow[d];
  lf *= 0.17677669529663687f;  // 1/sqrt(32)

  const float* qpn = qp + (size_t)n * (HC * PQC * 3) + h * (PQC * 3);
  const float* kpm = kp + (size_t)m * (HC * PQC * 3) + h * (PQC * 3);
  float lp = 0.0f;
#pragma unroll
  for (int p = 0; p < PQC; ++p) {
    float dx = qpn[p * 3 + 0] - kpm[p * 3 + 0];
    float dy = qpn[p * 3 + 1] - kpm[p * 3 + 1];
    float dz = qpn[p * 3 + 2] - kpm[p * 3 + 2];
    lp -= sqrtf(dx * dx + dy * dy + dz * dz + 1e-8f);
  }
  const float logit = w_c[h] * lf + w_l[h] * lp;
  wsm[m] = logit;
  __syncthreads();
  float mx = -3.4e38f;
  for (int t = 0; t < NRES; ++t) mx = fmaxf(mx, wsm[t]);
  const float e = __expf(logit - mx);
  __syncthreads();
  wsm[m] = e;
  __syncthreads();
  float s = 0.0f;
  for (int t = 0; t < NRES; ++t) s += wsm[t];
  const float w = e / s;
  __syncthreads();
  wsm[m] = w;
  __syncthreads();

  if (m < DH + PVC * 3) {  // 56 output dims per (n,h)
    float acc = 0.0f;
    if (m < DH) {
      for (int t = 0; t < NRES; ++t) acc += wsm[t] * V[(size_t)t * DIMC + h * DH + m];
      cat448h[(size_t)n * 448 + h * DH + m] = (half_t)acc;
    } else {
      const int e2 = m - DH;
      for (int t = 0; t < NRES; ++t)
        acc += wsm[t] * vp[(size_t)t * (HC * PVC * 3) + h * (PVC * 3) + e2];
      cat448h[(size_t)n * 448 + DIMC + h * (PVC * 3) + e2] = (half_t)acc;
    }
  }
}

// ---------------------------------------------------------------------------
// Fused EGNN edge pipeline. One block per residue i, 128 threads (4 waves).
// Streams j in tiles of 16: h0 = silu(Pi + Pj + dist*wd + b0)  (LDS f16)
//   eh = h0 @ We1 + b1           (WMMA 16x128x128, B from L2)
//   ch = silu(eh @ Wc0 + bc0)    (WMMA 16x128x128, B from L2)
//   cw = ch . wc1 + bc1;  agg += eh;  cacc += cw * (ci - cj)
// ---------------------------------------------------------------------------
__global__ void egnn_edge_kernel(const float* __restrict__ Pi, const float* __restrict__ Pj,
                                 const float* __restrict__ wd, const float* __restrict__ b0,
                                 const half_t* __restrict__ We1t, const float* __restrict__ b1,
                                 const half_t* __restrict__ Wc0t, const float* __restrict__ bc0,
                                 const float* __restrict__ wc1, const float* __restrict__ bc1,
                                 const float* __restrict__ coords, float* __restrict__ agg,
                                 float* __restrict__ coords_out) {
  const int i    = blockIdx.x;
  const int tid  = threadIdx.x;  // 128
  const int lane = tid & 31;
  const int wave = tid >> 5;     // 4 waves -> N chunks of 32

  __shared__ half_t h0h[16][HIDD];
  __shared__ float  ehf[16][HIDD];
  __shared__ half_t ehh[16][HIDD];
  __shared__ float  chf[16][HIDD];
  __shared__ float  cj[16][3];
  __shared__ float  distS[16];
  __shared__ float  cwj[16];
  __shared__ float  cacc[3];

  const float ci0 = coords[i * 3 + 0], ci1 = coords[i * 3 + 1], ci2 = coords[i * 3 + 2];
  const float pid = Pi[(size_t)i * HIDD + tid];
  const float wdd = wd[tid];
  const float b0d = b0[tid];
  float aggd = 0.0f;
  if (tid < 3) cacc[tid] = 0.0f;

  const int r    = lane & 15;
  const int koff = (lane >> 4) * 8;

  for (int jt = 0; jt < NRES / 16; ++jt) {
    __syncthreads();
    if (tid < 48) cj[tid / 3][tid % 3] = coords[(jt * 16 + tid / 3) * 3 + (tid % 3)];
    __syncthreads();
    if (tid < 16) {
      float dx = ci0 - cj[tid][0], dy = ci1 - cj[tid][1], dz = ci2 - cj[tid][2];
      distS[tid] = sqrtf(dx * dx + dy * dy + dz * dz + 1e-8f);
    }
    __syncthreads();
#pragma unroll 4
    for (int jj = 0; jj < 16; ++jj) {
      int j = jt * 16 + jj;
      float v = pid + Pj[(size_t)j * HIDD + tid] + distS[jj] * wdd + b0d;
      v = v / (1.0f + __expf(-v));  // silu
      h0h[jj][tid] = (half_t)v;
    }
    __syncthreads();

    // ---- e1: eh = h0 @ We1 + b1 ----
#pragma unroll
    for (int t = 0; t < 2; ++t) {
      const int ncol = wave * 32 + t * 16 + r;
      const half_t* brow = We1t + (size_t)ncol * HIDD;
      v8f acc = {};
#pragma unroll
      for (int ks = 0; ks < 4; ++ks) {
        v16h av, bv;
        ((uint4*)&av)[0] = *(const uint4*)&h0h[r][ks * 32 + koff];
        ((uint4*)&av)[1] = *(const uint4*)&h0h[r][ks * 32 + 16 + koff];
        ((uint4*)&bv)[0] = *(const uint4*)(brow + ks * 32 + koff);
        ((uint4*)&bv)[1] = *(const uint4*)(brow + ks * 32 + 16 + koff);
        acc = __builtin_amdgcn_wmma_f32_16x16x32_f16(false, av, false, bv, (short)0, acc,
                                                     false, false);
      }
      const float bb = b1[ncol];
#pragma unroll
      for (int v2 = 0; v2 < 8; ++v2) {
        int jj = v2 + ((lane >> 4) ? 8 : 0);
        float x = acc[v2] + bb;
        ehf[jj][ncol] = x;
        ehh[jj][ncol] = (half_t)x;
      }
    }
    __syncthreads();

#pragma unroll 4
    for (int jj = 0; jj < 16; ++jj) aggd += ehf[jj][tid];

    // ---- c0: ch = silu(eh @ Wc0 + bc0) ----
#pragma unroll
    for (int t = 0; t < 2; ++t) {
      const int ncol = wave * 32 + t * 16 + r;
      const half_t* brow = Wc0t + (size_t)ncol * HIDD;
      v8f acc = {};
#pragma unroll
      for (int ks = 0; ks < 4; ++ks) {
        v16h av, bv;
        ((uint4*)&av)[0] = *(const uint4*)&ehh[r][ks * 32 + koff];
        ((uint4*)&av)[1] = *(const uint4*)&ehh[r][ks * 32 + 16 + koff];
        ((uint4*)&bv)[0] = *(const uint4*)(brow + ks * 32 + koff);
        ((uint4*)&bv)[1] = *(const uint4*)(brow + ks * 32 + 16 + koff);
        acc = __builtin_amdgcn_wmma_f32_16x16x32_f16(false, av, false, bv, (short)0, acc,
                                                     false, false);
      }
      const float bb = bc0[ncol];
#pragma unroll
      for (int v2 = 0; v2 < 8; ++v2) {
        int jj = v2 + ((lane >> 4) ? 8 : 0);
        float x = acc[v2] + bb;
        chf[jj][ncol] = x / (1.0f + __expf(-x));
      }
    }
    __syncthreads();

    if (tid < 16) {
      float dot = bc1[0];
      for (int d = 0; d < HIDD; ++d) dot += chf[tid][d] * wc1[d];
      cwj[tid] = dot;
    }
    __syncthreads();
    if (tid < 48) {
      int jj = tid / 3, c = tid % 3;
      float cic = (c == 0) ? ci0 : (c == 1) ? ci1 : ci2;
      atomicAdd(&cacc[c], cwj[jj] * (cic - cj[jj][c]));
    }
  }
  __syncthreads();
  agg[(size_t)i * HIDD + tid] = aggd * (1.0f / NRES);
  if (tid < 3)
    coords_out[i * 3 + tid] = coords[i * 3 + tid] + 0.1f * cacc[tid] * (1.0f / NRES);
}

__global__ void copy_kernel(float* __restrict__ dst, const float* __restrict__ src, int n) {
  int i = blockIdx.x * 256 + threadIdx.x;
  if (i < n) dst[i] = src[i];
}

__global__ void copyconv_kernel(float* __restrict__ dst, half_t* __restrict__ dsth,
                                const float* __restrict__ src, int n) {
  int i = blockIdx.x * 256 + threadIdx.x;
  if (i < n) {
    float v = src[i];
    dst[i] = v;
    dsth[i] = (half_t)v;
  }
}

// f16 concat [feat(f32) | agg(f32)] -> cat384h, feeding the n0 GEMM.
__global__ void concat_kernel(half_t* __restrict__ dst, const float* __restrict__ a,
                              const float* __restrict__ b, int M, int Ka, int Kb) {
  int i = blockIdx.x * 256 + threadIdx.x;
  int Kt = Ka + Kb;
  if (i < M * Kt) {
    int r = i / Kt, c = i % Kt;
    float v = (c < Ka) ? a[(size_t)r * Ka + c] : b[(size_t)r * Kb + (c - Ka)];
    dst[i] = (half_t)v;
  }
}

__global__ void layernorm_kernel(float* __restrict__ x, half_t* __restrict__ xh,
                                 const float* __restrict__ g, const float* __restrict__ b) {
  const int n = blockIdx.x, d = threadIdx.x;
  __shared__ float red[DIMC];
  float v = x[(size_t)n * DIMC + d];
  red[d] = v;
  __syncthreads();
  for (int s = 128; s > 0; s >>= 1) {
    if (d < s) red[d] += red[d + s];
    __syncthreads();
  }
  const float mu = red[0] * (1.0f / DIMC);
  __syncthreads();
  const float c = v - mu;
  red[d] = c * c;
  __syncthreads();
  for (int s = 128; s > 0; s >>= 1) {
    if (d < s) red[d] += red[d + s];
    __syncthreads();
  }
  const float var = red[0] * (1.0f / DIMC);
  const float out = c * rsqrtf(var + 1e-5f) * g[d] + b[d];
  x[(size_t)n * DIMC + d] = out;
  xh[(size_t)n * DIMC + d] = (half_t)out;
}

// ---------------------------------------------------------------------------
// Host driver. Parameter order = jax tree_flatten (dict keys sorted):
//   0 features, 1 initial_coords,
//   2..9 heads: ch0.b ch0.w ch1.b ch1.w cf0.b cf0.w cf1.b cf1.w
//   per layer (base = 10 + 34*l), sub-dicts sorted egnn,ffn,ipa,norm:
//     egnn: c0.b c0.w c1.b c1.w e0.b e0.w e1.b e1.w n0.b n0.w n1.b n1.w
//     ffn:  f0.b f0.w f1.b f1.w
//     ipa:  k.b k.w kp.b kp.w out.b out.w q.b q.w qp.b qp.w v.b v.w vp.b vp.w w_c w_l
//     norm: b g
// ---------------------------------------------------------------------------
extern "C" void kernel_launch(void* const* d_in, const int* in_sizes, int n_in,
                              void* d_out, int out_size, void* d_ws, size_t ws_size,
                              hipStream_t stream) {
  (void)in_sizes; (void)n_in; (void)out_size; (void)ws_size;
  auto P = [&](int i) { return (const float*)d_in[i]; };

  const float* features    = P(0);
  const float* init_coords = P(1);
  const float *ch0_b = P(2), *ch0_w = P(3), *ch1_b = P(4), *ch1_w = P(5);
  const float *cf0_b = P(6), *cf0_w = P(7), *cf1_b = P(8), *cf1_w = P(9);

  // ---- f32 workspace ----
  float* ws = (float*)d_ws;
  float* feat = ws;             // 384*256
  float* cA   = feat + 98304;   // 384*3
  float* cB   = cA + 1152;
  float* Qb   = cB + 1152;      // 384*256
  float* Kb   = Qb + 98304;
  float* Vb   = Kb + 98304;
  float* qpb  = Vb + 98304;     // 384*96
  float* kpb  = qpb + 36864;
  float* vpb  = kpb + 36864;    // 384*192
  float* Pib  = vpb + 73728;    // 384*128
  float* Pjb  = Pib + 49152;
  float* aggb = Pjb + 49152;    // 384*128
  float* hid  = aggb + 49152;   // 384*1024 (largest intermediate)
  // ---- f16 workspace (starts 16B-aligned: 690432+393216=1083648 floats) ----
  half_t* hbase    = (half_t*)(hid + 393216);
  half_t* feat_h   = hbase;              // 384*256
  half_t* cat448_h = feat_h + 98304;     // 384*448
  half_t* cat384_h = cat448_h + 172032;  // 384*384
  half_t* hid_h    = cat384_h + 147456;  // 384*1024
  half_t* Wt_q   = hid_h + 393216;       // 256*256
  half_t* Wt_k   = Wt_q + 65536;
  half_t* Wt_v   = Wt_k + 65536;
  half_t* Wt_qp  = Wt_v + 65536;         // 96*256
  half_t* Wt_kp  = Wt_qp + 24576;
  half_t* Wt_vp  = Wt_kp + 24576;        // 192*256
  half_t* Wt_out = Wt_vp + 49152;        // 256*448
  half_t* Wt_ei  = Wt_out + 114688;      // 128*256
  half_t* Wt_ej  = Wt_ei + 32768;        // 128*256
  half_t* Wt_e1  = Wt_ej + 32768;        // 128*128
  half_t* Wt_c0  = Wt_e1 + 16384;        // 128*128
  half_t* Wt_n0  = Wt_c0 + 16384;        // 128*384
  half_t* Wt_n1  = Wt_n0 + 49152;        // 256*128
  half_t* Wt_f0  = Wt_n1 + 32768;        // 1024*256
  half_t* Wt_f1  = Wt_f0 + 262144;       // 256*1024
  half_t* Wt_ch0 = Wt_f1 + 262144;       // 256*256
  half_t* Wt_ch1 = Wt_ch0 + 65536;       // 32(pad 3)*256
  half_t* Wt_cf0 = Wt_ch1 + 8192;        // 128*256
  half_t* Wt_cf1 = Wt_cf0 + 32768;       // 32(pad 1)*128

  copyconv_kernel<<<(98304 + 255) / 256, 256, 0, stream>>>(feat, feat_h, features, 98304);
  copy_kernel<<<(1152 + 255) / 256, 256, 0, stream>>>(cA, init_coords, 1152);

  auto gemm = [&](const half_t* Ah, const half_t* Wt, const float* bias, const float* res,
                  float* C, half_t* Ch, int M, int Kd, int Nd, int act) {
    dim3 g((M + 63) / 64, (Nd + 31) / 32);
    gemm_kernel<<<g, 256, 0, stream>>>(Ah, Wt, bias, res, C, Ch, M, Kd, Nd, act);
  };
  auto trconv = [&](const float* W, half_t* Wt, int Kd, int Nd, int Npad) {
    trconv_kernel<<<(Npad * Kd + 255) / 256, 256, 0, stream>>>(W, Wt, Kd, Nd, Npad);
  };

  // Head weights: convert once.
  trconv(ch0_w, Wt_ch0, DIMC, DIMC, DIMC);
  trconv(ch1_w, Wt_ch1, DIMC, 3, 32);
  trconv(cf0_w, Wt_cf0, DIMC, HIDD, HIDD);
  trconv(cf1_w, Wt_cf1, HIDD, 1, 32);

  float* ccur = cA;
  float* cnext = cB;

  for (int l = 0; l < 2; ++l) {
    const int base = 10 + 34 * l;
    const float *c0b = P(base + 0),  *c0w = P(base + 1);
    const float *c1b = P(base + 2),  *c1w = P(base + 3);
    const float *e0b = P(base + 4),  *e0w = P(base + 5);
    const float *e1b = P(base + 6),  *e1w = P(base + 7);
    const float *n0b = P(base + 8),  *n0w = P(base + 9);
    const float *n1b = P(base + 10), *n1w = P(base + 11);
    const float *f0b = P(base + 12), *f0w = P(base + 13);
    const float *f1b = P(base + 14), *f1w = P(base + 15);
    const float *kbi  = P(base + 16), *kwi  = P(base + 17);
    const float *kpbi = P(base + 18), *kpwi = P(base + 19);
    const float *outb = P(base + 20), *outw = P(base + 21);
    const float *qbi  = P(base + 22), *qwi  = P(base + 23);
    const float *qpbi = P(base + 24), *qpwi = P(base + 25);
    const float *vbi  = P(base + 26), *vwi  = P(base + 27);
    const float *vpbi = P(base + 28), *vpwi = P(base + 29);
    const float *wcv = P(base + 30), *wlv = P(base + 31);
    const float *nrb = P(base + 32), *nrg = P(base + 33);

    // Per-layer weight conversion (f32 [K][N] -> f16 [Npad][K]).
    trconv(qwi,  Wt_q,  DIMC, DIMC, DIMC);
    trconv(kwi,  Wt_k,  DIMC, DIMC, DIMC);
    trconv(vwi,  Wt_v,  DIMC, DIMC, DIMC);
    trconv(qpwi, Wt_qp, DIMC, HC * PQC * 3, HC * PQC * 3);
    trconv(kpwi, Wt_kp, DIMC, HC * PQC * 3, HC * PQC * 3);
    trconv(vpwi, Wt_vp, DIMC, HC * PVC * 3, HC * PVC * 3);
    trconv(outw, Wt_out, DIMC + HC * PVC * 3, DIMC, DIMC);
    trconv(e0w,                            Wt_ei, DIMC, HIDD, HIDD);  // e0 rows [0,256)
    trconv(e0w + (size_t)DIMC * HIDD,      Wt_ej, DIMC, HIDD, HIDD);  // e0 rows [256,512)
    trconv(e1w, Wt_e1, HIDD, HIDD, HIDD);
    trconv(c0w, Wt_c0, HIDD, HIDD, HIDD);
    trconv(n0w, Wt_n0, DIMC + HIDD, HIDD, HIDD);
    trconv(n1w, Wt_n1, HIDD, DIMC, DIMC);
    trconv(f0w, Wt_f0, DIMC, 4 * DIMC, 4 * DIMC);
    trconv(f1w, Wt_f1, 4 * DIMC, DIMC, DIMC);

    // ---- IPA (frames are identity in the reference) ----
    gemm(feat_h, Wt_q,  qbi,  nullptr, Qb,  nullptr, NRES, DIMC, DIMC, ACT_NONE);
    gemm(feat_h, Wt_k,  kbi,  nullptr, Kb,  nullptr, NRES, DIMC, DIMC, ACT_NONE);
    gemm(feat_h, Wt_v,  vbi,  nullptr, Vb,  nullptr, NRES, DIMC, DIMC, ACT_NONE);
    gemm(feat_h, Wt_qp, qpbi, nullptr, qpb, nullptr, NRES, DIMC, HC * PQC * 3, ACT_NONE);
    gemm(feat_h, Wt_kp, kpbi, nullptr, kpb, nullptr, NRES, DIMC, HC * PQC * 3, ACT_NONE);
    gemm(feat_h, Wt_vp, vpbi, nullptr, vpb, nullptr, NRES, DIMC, HC * PVC * 3, ACT_NONE);
    ipa_attn_kernel<<<dim3(NRES, HC), 384, 0, stream>>>(Qb, Kb, Vb, qpb, kpb, vpb, wcv,
                                                        wlv, cat448_h);
    gemm(cat448_h, Wt_out, outb, feat, feat, feat_h, NRES, DIMC + HC * PVC * 3, DIMC,
         ACT_NONE);

    // ---- EGNN ----
    gemm(feat_h, Wt_ei, nullptr, nullptr, Pib, nullptr, NRES, DIMC, HIDD, ACT_NONE);
    gemm(feat_h, Wt_ej, nullptr, nullptr, Pjb, nullptr, NRES, DIMC, HIDD, ACT_NONE);
    egnn_edge_kernel<<<NRES, HIDD, 0, stream>>>(
        Pib, Pjb, e0w + (size_t)2 * DIMC * HIDD, e0b, Wt_e1, e1b, Wt_c0, c0b, c1w, c1b,
        ccur, aggb, cnext);
    concat_kernel<<<(NRES * (DIMC + HIDD) + 255) / 256, 256, 0, stream>>>(
        cat384_h, feat, aggb, NRES, DIMC, HIDD);
    gemm(cat384_h, Wt_n0, n0b, nullptr, hid,  hid_h,  NRES, DIMC + HIDD, HIDD, ACT_SILU);
    gemm(hid_h,    Wt_n1, n1b, feat,    feat, feat_h, NRES, HIDD, DIMC, ACT_NONE);

    layernorm_kernel<<<NRES, DIMC, 0, stream>>>(feat, feat_h, nrg, nrb);

    // ---- FFN ----
    gemm(feat_h, Wt_f0, f0b, nullptr, hid,  hid_h,  NRES, DIMC, 4 * DIMC, ACT_GELU);
    gemm(hid_h,  Wt_f1, f1b, feat,    feat, feat_h, NRES, 4 * DIMC, DIMC, ACT_NONE);

    float* t = ccur; ccur = cnext; cnext = t;
  }

  // ---- Heads ----
  float* outp = (float*)d_out;
  gemm(feat_h, Wt_ch0, ch0_b, nullptr, hid,  hid_h, NRES, DIMC, DIMC, ACT_GELU);
  gemm(hid_h,  Wt_ch1, ch1_b, ccur,    outp, nullptr, NRES, DIMC, 3, ACT_NONE);
  gemm(feat_h, Wt_cf0, cf0_b, nullptr, hid,  hid_h, NRES, DIMC, HIDD, ACT_GELU);
  gemm(hid_h,  Wt_cf1, cf1_b, nullptr, outp + NRES * 3, nullptr, NRES, HIDD, 1,
       ACT_SIGMOID);
}
